// MoELayer_66116726554791
// MI455X (gfx1250) — matrix operand order repaired and checked by reading, not compiled, not run on hardware
//
#include <hip/hip_runtime.h>
#include <hip/hip_bf16.h>

// MoE: B=8 S=2048 D=512 F=2048 E=8, top-2 routing.
// Strategy: route tokens (4x less work than dense-8-expert reference math),
// bf16 WMMA (v_wmma_f32_16x16x32_bf16) for both expert GEMMs, fused through LDS.

#define BB 8
#define SS 2048
#define DD 512
#define FF 2048
#define EE 8
#define NTOK (BB * SS)      // 16384 tokens
#define MTILE 32            // tokens per FFN workgroup tile (2x 16-row WMMA)
#define FCH 256             // F-chunk staged in LDS between GEMM1 and GEMM2

typedef __attribute__((ext_vector_type(16))) __bf16 bf16x16;
typedef __attribute__((ext_vector_type(8)))  float  f32x8;

union Frag { bf16x16 v; float4 q[2]; };

// ---------------- workspace layout (bytes) ----------------
constexpr size_t OFF_GATESUM = 0;                                   // 8 f32
constexpr size_t OFF_CNT     = 256;                                 // 8 i32
constexpr size_t OFF_TOK     = 512;                                 // EE*NTOK i32
constexpr size_t OFF_WGT     = OFF_TOK + (size_t)EE * NTOK * 4;     // EE*NTOK f32
constexpr size_t OFF_XBF     = (OFF_WGT + (size_t)EE * NTOK * 4 + 255) & ~(size_t)255;
constexpr size_t OFF_W1T     = OFF_XBF + (size_t)NTOK * DD * 2;     // [E][F][D] bf16
constexpr size_t OFF_W2T     = OFF_W1T + (size_t)EE * DD * FF * 2;  // [E][D][F] bf16
// total ~49 MB of d_ws

// ---------------- init ----------------
__global__ void zero_kernel(float* __restrict__ out, size_t n,
                            float* __restrict__ gateSum, int* __restrict__ cnt) {
  size_t i = (size_t)blockIdx.x * blockDim.x + threadIdx.x;
  size_t stride = (size_t)gridDim.x * blockDim.x;
  for (size_t j = i; j < n; j += stride) out[j] = 0.0f;
  if (i < EE) { gateSum[i] = 0.0f; cnt[i] = 0; }
}

// ---------------- f32 -> bf16 straight convert (x) ----------------
__global__ void cvt_x_kernel(const float* __restrict__ in, __bf16* __restrict__ outp, size_t n) {
  size_t i = (size_t)blockIdx.x * blockDim.x + threadIdx.x;
  size_t stride = (size_t)gridDim.x * blockDim.x;
  for (size_t j = i; j < n; j += stride) outp[j] = (__bf16)in[j];
}

// ---------------- per-expert tiled transpose + convert: in [R][C] f32 -> out [C][R] bf16 ----------------
__global__ void transpose_cvt_kernel(const float* __restrict__ in, __bf16* __restrict__ outp,
                                     int R, int C) {
  __shared__ float tile[32][33];
  const size_t base = (size_t)blockIdx.z * (size_t)R * C;
  const int r0 = blockIdx.y * 32;
  const int c0 = blockIdx.x * 32;
  const int tx = threadIdx.x & 31;
  const int ty = threadIdx.x >> 5;   // 8 row-groups
  #pragma unroll
  for (int rr = ty; rr < 32; rr += 8)
    tile[rr][tx] = in[base + (size_t)(r0 + rr) * C + c0 + tx];
  __syncthreads();
  #pragma unroll
  for (int rr = ty; rr < 32; rr += 8)
    outp[base + (size_t)(c0 + rr) * R + r0 + tx] = (__bf16)tile[tx][rr];
}

// ---------------- router: logits -> softmax -> top-2 -> expert lists ----------------
__global__ __launch_bounds__(256) void router_kernel(
    const float* __restrict__ x, const float* __restrict__ Wr, const float* __restrict__ br,
    float* __restrict__ gateSum, int* __restrict__ cnt,
    int* __restrict__ tokList, float* __restrict__ wList) {
  __shared__ float WrS[DD * EE];     // 16 KB
  __shared__ float gAcc[EE];
  const int t = threadIdx.x;
  for (int i = t; i < DD * EE; i += blockDim.x) WrS[i] = Wr[i];
  if (t < EE) gAcc[t] = 0.0f;
  __syncthreads();

  const int tok = blockIdx.x * blockDim.x + t;
  float acc[EE];
  #pragma unroll
  for (int e = 0; e < EE; ++e) acc[e] = br[e];
  const float* xr = x + (size_t)tok * DD;
  for (int d = 0; d < DD; ++d) {
    const float xv = xr[d];
    #pragma unroll
    for (int e = 0; e < EE; ++e) acc[e] += xv * WrS[d * EE + e];
  }
  // softmax over E=8
  float m = acc[0];
  #pragma unroll
  for (int e = 1; e < EE; ++e) m = fmaxf(m, acc[e]);
  float s = 0.0f;
  #pragma unroll
  for (int e = 0; e < EE; ++e) { acc[e] = __expf(acc[e] - m); s += acc[e]; }
  const float inv = 1.0f / s;
  #pragma unroll
  for (int e = 0; e < EE; ++e) acc[e] *= inv;
  // top-2 (ties -> lowest index, matching lax.top_k)
  int i1 = 0; float v1 = acc[0];
  #pragma unroll
  for (int e = 1; e < EE; ++e) if (acc[e] > v1) { v1 = acc[e]; i1 = e; }
  int i2 = (i1 == 0) ? 1 : 0; float v2 = acc[i2];
  #pragma unroll
  for (int e = 0; e < EE; ++e) if (e != i1 && acc[e] > v2) { v2 = acc[e]; i2 = e; }

  int p1 = atomicAdd(&cnt[i1], 1);
  tokList[(size_t)i1 * NTOK + p1] = tok;  wList[(size_t)i1 * NTOK + p1] = v1;
  int p2 = atomicAdd(&cnt[i2], 1);
  tokList[(size_t)i2 * NTOK + p2] = tok;  wList[(size_t)i2 * NTOK + p2] = v2;

  #pragma unroll
  for (int e = 0; e < EE; ++e) atomicAdd(&gAcc[e], acc[e]);
  __syncthreads();
  if (t < EE) atomicAdd(&gateSum[t], gAcc[t]);
}

// ---------------- aux loss ----------------
__global__ void aux_kernel(const float* __restrict__ gateSum, float* __restrict__ out_loss) {
  if (threadIdx.x == 0) {
    float l = 0.0f;
    #pragma unroll
    for (int e = 0; e < EE; ++e) {
      float p = gateSum[e] * (1.0f / (float)NTOK);
      l += p * logf(p + 1e-9f);
    }
    *out_loss = l;
  }
}

// ---------------- fused expert FFN: relu(X@W1+b1)@W2+b2, gate-scaled scatter ----------------
__global__ __launch_bounds__(256) void moe_ffn_kernel(
    const __bf16* __restrict__ xbf, const __bf16* __restrict__ w1t,
    const float* __restrict__ b1, const __bf16* __restrict__ w2t,
    const float* __restrict__ b2, const int* __restrict__ cnt,
    const int* __restrict__ tokList, const float* __restrict__ wList,
    float* __restrict__ out) {
  const int e = blockIdx.y;
  const int tile = blockIdx.x;
  const int n_e = cnt[e];
  if (tile * MTILE >= n_e) return;

  extern __shared__ char smem[];
  __bf16* Xs = (__bf16*)smem;                                        // [32][512] bf16
  __bf16* Hs = (__bf16*)(smem + (size_t)MTILE * DD * 2);             // [32][FCH] bf16
  float*  Wg = (float*)(smem + (size_t)MTILE * DD * 2 + (size_t)MTILE * FCH * 2);
  int*    Tk = (int*)((char*)Wg + MTILE * 4);

  const int tid = threadIdx.x;
  if (tid < MTILE) {
    int idx = tile * MTILE + tid;
    if (idx < n_e) { Tk[tid] = tokList[(size_t)e * NTOK + idx]; Wg[tid] = wList[(size_t)e * NTOK + idx]; }
    else           { Tk[tid] = -1; Wg[tid] = 0.0f; }
  }
  __syncthreads();

  { // gather 32 token rows of bf16 x into LDS (1 KB per row, 8 threads/row)
    const int row = tid >> 3, sub = tid & 7;
    const int t = Tk[row];
    uint4* dst = (uint4*)(Xs + (size_t)row * DD);
    if (t >= 0) {
      const uint4* src = (const uint4*)(xbf + (size_t)t * DD);
      #pragma unroll
      for (int j = 0; j < 8; ++j) dst[sub * 8 + j] = src[sub * 8 + j];
    } else {
      uint4 z; z.x = z.y = z.z = z.w = 0u;
      #pragma unroll
      for (int j = 0; j < 8; ++j) dst[sub * 8 + j] = z;
    }
  }
  __syncthreads();

  const int lane  = tid & 31;
  const int wave  = tid >> 5;            // 8 waves
  const int ln    = lane & 15;           // M/N index within 16
  const int khA   = (lane & 16) ? 8 : 0; // 16-bit A-matrix K-half offset
  const int khB   = (lane & 16) ? 16 : 0;// unused; B loads use khA
  const int mbase = (lane & 16) ? 8 : 0; // f32 C/D: VGPR j -> M = mbase + j

  const float*  b1g = b1  + (size_t)e * FF;
  const float*  b2g = b2  + (size_t)e * DD;
  const __bf16* w1e = w1t + (size_t)e * FF * DD;   // K-contiguous: [F][D]
  const __bf16* w2e = w2t + (size_t)e * DD * FF;   // K-contiguous: [D][F]

  // persistent GEMM2 accumulators: 4 output n-tiles per wave x 2 m-subtiles
  f32x8 acc2[4][2];
  #pragma unroll
  for (int i = 0; i < 4; ++i) {
    const int n0 = (wave * 4 + i) * 16;
    const float b = b2g[n0 + ln];
    #pragma unroll
    for (int j = 0; j < 8; ++j) { acc2[i][0][j] = b; acc2[i][1][j] = b; }
  }

  for (int kc = 0; kc < FF; kc += FCH) {
    // ---- GEMM1 chunk: Hs = relu(X @ W1[:, kc:kc+FCH] + b1) ----
    for (int nt = wave; nt < FCH / 16; nt += 8) {
      const int nG = kc + nt * 16 + ln;
      const float b = b1g[nG];
      f32x8 acc0, acc1;
      #pragma unroll
      for (int j = 0; j < 8; ++j) { acc0[j] = b; acc1[j] = b; }
      const __bf16* Bcol = w1e + (size_t)nG * DD;
      const __bf16* Ar0  = Xs + (size_t)ln * DD;
      const __bf16* Ar1  = Xs + (size_t)(16 + ln) * DD;
      #pragma unroll 4
      for (int k0 = 0; k0 < DD; k0 += 32) {
        Frag bf, a0, a1;
        bf.q[0] = *(const float4*)(Bcol + k0 + khA);
        bf.q[1] = *(const float4*)(Bcol + k0 + 16 + khA);
        __builtin_prefetch(Bcol + k0 + 64, 0, 1);
        a0.q[0] = *(const float4*)(Ar0 + k0 + khA);
        a0.q[1] = *(const float4*)(Ar0 + k0 + 16 + khA);
        a1.q[0] = *(const float4*)(Ar1 + k0 + khA);
        a1.q[1] = *(const float4*)(Ar1 + k0 + 16 + khA);
        acc0 = __builtin_amdgcn_wmma_f32_16x16x32_bf16(false, a0.v, false, bf.v, (short)0, acc0, false, false);
        acc1 = __builtin_amdgcn_wmma_f32_16x16x32_bf16(false, a1.v, false, bf.v, (short)0, acc1, false, false);
      }
      const int c = nt * 16 + ln;
      #pragma unroll
      for (int j = 0; j < 8; ++j) {
        float h0 = acc0[j] > 0.0f ? acc0[j] : 0.0f;
        float h1 = acc1[j] > 0.0f ? acc1[j] : 0.0f;
        Hs[(size_t)(mbase + j) * FCH + c]      = (__bf16)h0;
        Hs[(size_t)(16 + mbase + j) * FCH + c] = (__bf16)h1;
      }
    }
    __syncthreads();

    // ---- GEMM2 partial: acc2 += Hs @ W2[kc:kc+FCH, :] ----
    #pragma unroll
    for (int i = 0; i < 4; ++i) {
      const int n0 = (wave * 4 + i) * 16;
      const __bf16* Bcol = w2e + (size_t)(n0 + ln) * FF + kc;
      const __bf16* Ar0  = Hs + (size_t)ln * FCH;
      const __bf16* Ar1  = Hs + (size_t)(16 + ln) * FCH;
      #pragma unroll 4
      for (int k0 = 0; k0 < FCH; k0 += 32) {
        Frag bf, a0, a1;
        bf.q[0] = *(const float4*)(Bcol + k0 + khA);
        bf.q[1] = *(const float4*)(Bcol + k0 + 16 + khA);
        __builtin_prefetch(Bcol + k0 + 64, 0, 1);
        a0.q[0] = *(const float4*)(Ar0 + k0 + khA);
        a0.q[1] = *(const float4*)(Ar0 + k0 + 16 + khA);
        a1.q[0] = *(const float4*)(Ar1 + k0 + khA);
        a1.q[1] = *(const float4*)(Ar1 + k0 + 16 + khA);
        acc2[i][0] = __builtin_amdgcn_wmma_f32_16x16x32_bf16(false, a0.v, false, bf.v, (short)0, acc2[i][0], false, false);
        acc2[i][1] = __builtin_amdgcn_wmma_f32_16x16x32_bf16(false, a1.v, false, bf.v, (short)0, acc2[i][1], false, false);
      }
    }
    __syncthreads();
  }

  // ---- epilogue: out[tok, n] += gate * y ----
  #pragma unroll
  for (int i = 0; i < 4; ++i) {
    const int n0 = (wave * 4 + i) * 16;
    #pragma unroll
    for (int j = 0; j < 8; ++j) {
      const int m0 = mbase + j, m1 = 16 + mbase + j;
      const int t0 = Tk[m0], t1 = Tk[m1];
      if (t0 >= 0) atomicAdd(out + (size_t)t0 * DD + n0 + ln, Wg[m0] * acc2[i][0][j]);
      if (t1 >= 0) atomicAdd(out + (size_t)t1 * DD + n0 + ln, Wg[m1] * acc2[i][1][j]);
    }
  }
}

extern "C" void kernel_launch(void* const* d_in, const int* in_sizes, int n_in,
                              void* d_out, int out_size, void* d_ws, size_t ws_size,
                              hipStream_t stream) {
  const float* x  = (const float*)d_in[0];
  const float* Wr = (const float*)d_in[1];
  const float* br = (const float*)d_in[2];
  const float* W1 = (const float*)d_in[3];
  const float* b1 = (const float*)d_in[4];
  const float* W2 = (const float*)d_in[5];
  const float* b2 = (const float*)d_in[6];
  float* out = (float*)d_out;          // [B*S*D] out + [1] aux_loss

  char* ws = (char*)d_ws;
  float*  gateSum = (float*)(ws + OFF_GATESUM);
  int*    cnt     = (int*)(ws + OFF_CNT);
  int*    tokList = (int*)(ws + OFF_TOK);
  float*  wList   = (float*)(ws + OFF_WGT);
  __bf16* xbf     = (__bf16*)(ws + OFF_XBF);
  __bf16* w1t     = (__bf16*)(ws + OFF_W1T);
  __bf16* w2t     = (__bf16*)(ws + OFF_W2T);

  const size_t n_out_main = (size_t)NTOK * DD;

  zero_kernel<<<2048, 256, 0, stream>>>(out, n_out_main + 1, gateSum, cnt);
  cvt_x_kernel<<<4096, 256, 0, stream>>>(x, xbf, n_out_main);
  transpose_cvt_kernel<<<dim3(FF / 32, DD / 32, EE), 256, 0, stream>>>(W1, w1t, DD, FF);
  transpose_cvt_kernel<<<dim3(DD / 32, FF / 32, EE), 256, 0, stream>>>(W2, w2t, FF, DD);
  router_kernel<<<NTOK / 256, 256, 0, stream>>>(x, Wr, br, gateSum, cnt, tokList, wList);
  aux_kernel<<<1, 64, 0, stream>>>(gateSum, out + n_out_main);

  const size_t smem = (size_t)MTILE * DD * 2 + (size_t)MTILE * FCH * 2 + MTILE * 4 + MTILE * 4; // 49408 B
  moe_ffn_kernel<<<dim3(NTOK / MTILE, EE), 256, smem, stream>>>(
      xbf, w1t, b1, w2t, b2, cnt, tokList, wList, out);
}
